// myLSTM_88983132438943
// MI455X (gfx1250) — matrix operand order repaired and checked by reading, not compiled, not run on hardware
//
#include <hip/hip_runtime.h>
#include <hip/hip_bf16.h>
#include <math.h>

// ---------------------------------------------------------------------------
// Problem constants (from reference): B=32, T=96, FEAT=2048, IN=4096
// ---------------------------------------------------------------------------
#define BB    32
#define TT    96
#define FEAT  2048
#define INF   4096
#define G4    (4 * FEAT)      // 8192
#define MROWS (TT * BB)       // 3072

typedef __attribute__((ext_vector_type(16))) __bf16 bf16x16;
typedef __attribute__((ext_vector_type(8)))  __bf16 bf16x8;
typedef __attribute__((ext_vector_type(8)))  float  f32x8;

// ---------------------------------------------------------------------------
// Helpers
// ---------------------------------------------------------------------------
__device__ __forceinline__ unsigned short f2bf_rne(float x) {
    unsigned u = __float_as_uint(x);
    unsigned r = u + 0x7FFFu + ((u >> 16) & 1u);
    return (unsigned short)(r >> 16);
}

__device__ __forceinline__ float sigm(float x) { return 1.0f / (1.0f + __expf(-x)); }

// A fragment: 16x32 (MxK) bf16, row-major source with leading dim lda.
// lanes 0-15 -> M=lane, K = 0..15; lanes 16-31 -> M=lane-16,
// K = 16..31 (matches load_b_frag).  Two contiguous 8-element (16B) vector loads per lane.
__device__ __forceinline__ bf16x16 load_a_frag(const __bf16* A, int lda,
                                               int m0, int k0, int lane) {
    int m  = m0 + (lane & 15);
    int hb = (lane >> 4) * 16;
    const __bf16* p = A + (size_t)m * lda + k0;
    bf16x8 lo = *(const bf16x8*)(p + hb);
    bf16x8 hi = *(const bf16x8*)(p + 8 + hb);
    return __builtin_shufflevector(lo, hi, 0, 1, 2, 3, 4, 5, 6, 7,
                                           8, 9, 10, 11, 12, 13, 14, 15);
}

// B fragment: 32x16 (KxN) bf16 where B[k][n] = W[n][k], W row-major [N,K] ld=ldw.
// lanes 0-15 -> N=lane, K=0..15; lanes 16-31 -> N=lane-16, K=16..31.
// 16 contiguous K per lane == one 32-byte vector load from W's row n.
__device__ __forceinline__ bf16x16 load_b_frag(const __bf16* W, int ldw,
                                               int n0, int k0, int lane) {
    int n  = n0 + (lane & 15);
    int kb = (lane >> 4) * 16;
    return *(const bf16x16*)(W + (size_t)n * ldw + k0 + kb);
}

__device__ __forceinline__ f32x8 wmma_bf16(bf16x16 a, bf16x16 b, f32x8 c) {
    return __builtin_amdgcn_wmma_f32_16x16x32_bf16(
        /*neg_a=*/false, a, /*neg_b=*/false, b,
        /*c_mod=*/(short)0, c, /*reuse_a=*/false, /*reuse_b=*/false);
}

// ---------------------------------------------------------------------------
// Elementwise conversion / init kernels
// ---------------------------------------------------------------------------
__global__ __launch_bounds__(256) void cvt_bf16_kernel(const float* __restrict__ src,
                                                       unsigned short* __restrict__ dst,
                                                       long n) {
    long i = (long)blockIdx.x * 256 + threadIdx.x;
    if (i < n) dst[i] = f2bf_rne(src[i]);
}

// feature [B, T, IN] -> Xbf [T*B, IN] (bf16), transposed to time-major
__global__ __launch_bounds__(256) void cvt_feature_kernel(const float* __restrict__ feat,
                                                          unsigned short* __restrict__ dst) {
    long i = (long)blockIdx.x * 256 + threadIdx.x;
    long n = (long)MROWS * INF;
    if (i >= n) return;
    int k  = (int)(i % INF);
    long r = i / INF;
    int b  = (int)(r % BB);
    int t  = (int)(r / BB);
    float v = feat[((long)b * TT + t) * INF + k];
    dst[i] = f2bf_rne(v);
}

// zero h2 (f32 + bf16) and c
__global__ __launch_bounds__(256) void init_state_kernel(float* __restrict__ h2f,
                                                         float* __restrict__ c,
                                                         unsigned short* __restrict__ h2bf) {
    int i = blockIdx.x * 256 + threadIdx.x;
    if (i < BB * FEAT) { h2f[i] = 0.0f; c[i] = 0.0f; h2bf[i] = 0; }
}

// ---------------------------------------------------------------------------
// Big input-projection GEMM:  Xg[3072, 8192] = Xbf[3072,4096] @ Wih[8192,4096]^T + b_ih
// Block = 256 threads = 8 waves; each wave owns a 32M x 64N register tile:
// 2 A fragments x 4 B fragments -> 8 WMMAs per K-step (~21 FLOP/byte from cache).
// ---------------------------------------------------------------------------
__global__ __launch_bounds__(256) void gemm_xw_kernel(const unsigned short* __restrict__ Xus,
                                                      const unsigned short* __restrict__ Wus,
                                                      const float* __restrict__ bias,
                                                      float* __restrict__ Y) {
    const __bf16* X = (const __bf16*)Xus;
    const __bf16* W = (const __bf16*)Wus;
    int lane  = threadIdx.x & 31;
    int wave  = threadIdx.x >> 5;
    int m0    = (blockIdx.x * 8 + wave) * 32;   // 12 * 8 = 96 super-tiles of 32 rows
    int n0    = blockIdx.y * 64;                // 128 N-strips

    f32x8 zero = {0.f, 0.f, 0.f, 0.f, 0.f, 0.f, 0.f, 0.f};
    f32x8 acc[2][4] = {{zero, zero, zero, zero}, {zero, zero, zero, zero}};

    for (int k0 = 0; k0 < INF; k0 += 32) {
        bf16x16 a0 = load_a_frag(X, INF, m0,      k0, lane);
        bf16x16 a1 = load_a_frag(X, INF, m0 + 16, k0, lane);
        // prefetch next K block of the weight stream (L2-resident reuse)
        if (k0 + 32 < INF)
            __builtin_prefetch(W + (size_t)(n0 + (lane & 15)) * INF + k0 + 32, 0, 3);
#pragma unroll
        for (int j = 0; j < 4; ++j) {
            bf16x16 b = load_b_frag(W, INF, n0 + 16 * j, k0, lane);
            acc[0][j] = wmma_bf16(a0, b, acc[0][j]);
            acc[1][j] = wmma_bf16(a1, b, acc[1][j]);
        }
    }

    int nl = lane & 15;
    int mh = (lane >> 4) * 8;
#pragma unroll
    for (int j = 0; j < 4; ++j) {
        int n = n0 + 16 * j + nl;
        float bv = bias[n];
#pragma unroll
        for (int i = 0; i < 2; ++i) {
#pragma unroll
            for (int v = 0; v < 8; ++v)
                Y[(size_t)(m0 + 16 * i + mh + v) * G4 + n] = acc[i][j][v] + bv;
        }
    }
}

// ---------------------------------------------------------------------------
// Recurrent step: gates = Xg[t] + h2 @ W_hh^T + b_hh ; c,h update.
// One block per 16 features of F: 8 waves = (4 gates) x (2 M-tiles of 16 rows).
// Gate tiles land in LDS so the same block does the i/f/g/o pointwise combine.
// Dominant traffic: W_hh_bf16 (32 MB) streamed once per step from the 192 MB L2.
// ---------------------------------------------------------------------------
__global__ __launch_bounds__(256) void step_gemm_kernel(const unsigned short* __restrict__ Hus,
                                                        const unsigned short* __restrict__ Wus,
                                                        const float* __restrict__ bhh,
                                                        const float* __restrict__ xg,
                                                        float* __restrict__ c,
                                                        float* __restrict__ hnew) {
    __shared__ float lds[4][BB][16];
    const __bf16* H = (const __bf16*)Hus;
    const __bf16* W = (const __bf16*)Wus;
    int lane = threadIdx.x & 31;
    int wave = threadIdx.x >> 5;
    int gate = wave >> 1;            // 0=i 1=f 2=g 3=o
    int m0   = (wave & 1) * 16;
    int nF   = blockIdx.x * 16;      // feature column within F (128 blocks)
    int nG   = gate * FEAT + nF;     // column within 4F

    f32x8 acc = {0.f, 0.f, 0.f, 0.f, 0.f, 0.f, 0.f, 0.f};
    for (int k0 = 0; k0 < FEAT; k0 += 32) {
        bf16x16 a = load_a_frag(H, FEAT, m0, k0, lane);
        bf16x16 b = load_b_frag(W, FEAT, nG, k0, lane);
        acc = wmma_bf16(a, b, acc);
    }

    int nl = lane & 15;
    int mh = (lane >> 4) * 8;
    float bb = bhh[nG + nl];
#pragma unroll
    for (int v = 0; v < 8; ++v) {
        int m = m0 + mh + v;
        lds[gate][m][nl] = acc[v] + bb + xg[(size_t)m * G4 + nG + nl];
    }
    __syncthreads();

    for (int e = threadIdx.x; e < BB * 16; e += 256) {
        int m = e >> 4, j = e & 15;
        float gi = lds[0][m][j];
        float gf = lds[1][m][j];
        float gg = lds[2][m][j];
        float go = lds[3][m][j];
        size_t off = (size_t)m * FEAT + nF + j;
        float c2 = sigm(gf) * c[off] + sigm(gi) * tanhf(gg);
        c[off]    = c2;
        hnew[off] = sigm(go) * tanhf(c2);
    }
}

// ---------------------------------------------------------------------------
// Gumbel-softmax gating + heads.  One block per batch row (32 blocks).
// t==0: no gating (h2 = h_new), matching the reference.
// ---------------------------------------------------------------------------
__global__ __launch_bounds__(256) void gate_heads_kernel(const float* __restrict__ hprev,
                                                         const float* __restrict__ hnew,
                                                         const float* __restrict__ Wuse,
                                                         const float* __restrict__ buse,
                                                         const float* __restrict__ Wpred,
                                                         const float* __restrict__ bpred,
                                                         const float* __restrict__ Wutil,
                                                         const float* __restrict__ butil,
                                                         const float* __restrict__ gu,
                                                         int t,
                                                         float* __restrict__ h2f,
                                                         unsigned short* __restrict__ h2bf,
                                                         float* __restrict__ pred,
                                                         float* __restrict__ util) {
    __shared__ float r0[256], r1[256], r2[256];
    __shared__ float sA0, sA1;
    int m = blockIdx.x, tid = threadIdx.x;
    const float* hn = hnew + (size_t)m * FEAT;
    const float* hp = hprev + (size_t)m * FEAT;

    float a0 = 0.0f, a1 = 1.0f;
    if (t > 0) {
        float l0 = 0.f, l1 = 0.f;
        for (int j = tid; j < FEAT; j += 256) {
            float v = hn[j];
            l0 += v * Wuse[j];
            l1 += v * Wuse[FEAT + j];
        }
        r0[tid] = l0; r1[tid] = l1;
        __syncthreads();
        for (int s = 128; s > 0; s >>= 1) {
            if (tid < s) { r0[tid] += r0[tid + s]; r1[tid] += r1[tid + s]; }
            __syncthreads();
        }
        if (tid == 0) {
            const float eps = 1e-10f;
            float u0 = gu[(((size_t)(t - 1)) * BB + m) * 2 + 0];
            float u1 = gu[(((size_t)(t - 1)) * BB + m) * 2 + 1];
            float g0 = -logf(-logf(u0 + eps) + eps);
            float g1 = -logf(-logf(u1 + eps) + eps);
            float z0 = r0[0] + buse[0] + g0;
            float z1 = r1[0] + buse[1] + g1;
            float mx = fmaxf(z0, z1);
            float e0 = __expf(z0 - mx), e1 = __expf(z1 - mx);
            float inv = 1.0f / (e0 + e1);
            sA0 = e0 * inv; sA1 = e1 * inv;
        }
        __syncthreads();
        a0 = sA0; a1 = sA1;
    }

    float p0 = 0.f, p1 = 0.f, uu = 0.f;
    for (int j = tid; j < FEAT; j += 256) {
        float h2 = a0 * hp[j] + a1 * hn[j];
        h2f [(size_t)m * FEAT + j] = h2;
        h2bf[(size_t)m * FEAT + j] = f2bf_rne(h2);
        p0 += h2 * Wpred[j];
        p1 += h2 * Wpred[FEAT + j];
        uu += h2 * Wutil[j];
    }
    r0[tid] = p0; r1[tid] = p1; r2[tid] = uu;
    __syncthreads();
    for (int s = 128; s > 0; s >>= 1) {
        if (tid < s) { r0[tid] += r0[tid + s]; r1[tid] += r1[tid + s]; r2[tid] += r2[tid + s]; }
        __syncthreads();
    }
    if (tid == 0) {
        pred[((size_t)t * BB + m) * 2 + 0] = r0[0] + bpred[0];
        pred[((size_t)t * BB + m) * 2 + 1] = r1[0] + bpred[1];
        util[(size_t)t * BB + m]           = r2[0] + butil[0];
    }
}

// ---------------------------------------------------------------------------
// Host-side launch
// ---------------------------------------------------------------------------
extern "C" void kernel_launch(void* const* d_in, const int* in_sizes, int n_in,
                              void* d_out, int out_size, void* d_ws, size_t ws_size,
                              hipStream_t stream) {
    (void)in_sizes; (void)n_in; (void)out_size; (void)ws_size;

    const float* feature  = (const float*)d_in[0];   // [B, T, IN]
    const float* gumbel_u = (const float*)d_in[1];   // [T-1, B, 2]
    const float* W_ih     = (const float*)d_in[2];   // [4F, IN]
    const float* b_ih     = (const float*)d_in[3];   // [4F]
    const float* W_hh     = (const float*)d_in[4];   // [4F, F]
    const float* b_hh     = (const float*)d_in[5];   // [4F]
    const float* W_pred   = (const float*)d_in[6];   // [2, F]
    const float* b_pred   = (const float*)d_in[7];   // [2]
    const float* W_util   = (const float*)d_in[8];   // [1, F]
    const float* b_util   = (const float*)d_in[9];   // [1]
    const float* W_use    = (const float*)d_in[10];  // [2, F]
    const float* b_use    = (const float*)d_in[11];  // [2]

    float* pred_out = (float*)d_out;                 // [T, B, 2]
    float* util_out = pred_out + (size_t)TT * BB * 2;// [T, B, 1]

    // --- workspace layout (256B aligned) ---
    char* base = (char*)d_ws;
    size_t off = 0;
    auto take = [&](size_t bytes) { char* p = base + off;
                                    off = (off + bytes + 255) & ~(size_t)255; return p; };
    unsigned short* Wih_bf = (unsigned short*)take((size_t)G4 * INF * 2);   // 64 MB
    unsigned short* Whh_bf = (unsigned short*)take((size_t)G4 * FEAT * 2);  // 32 MB
    unsigned short* Xbf    = (unsigned short*)take((size_t)MROWS * INF * 2);// 24 MB
    float*          Xg     = (float*)take((size_t)MROWS * G4 * 4);          // 96 MB
    float*          h2f    = (float*)take((size_t)BB * FEAT * 4);
    float*          hnew   = (float*)take((size_t)BB * FEAT * 4);
    float*          cbuf   = (float*)take((size_t)BB * FEAT * 4);
    unsigned short* h2bf   = (unsigned short*)take((size_t)BB * FEAT * 2);

    // 1) precision conversion + init
    {
        long n1 = (long)G4 * INF;
        cvt_bf16_kernel<<<(unsigned)((n1 + 255) / 256), 256, 0, stream>>>(W_ih, Wih_bf, n1);
        long n2 = (long)G4 * FEAT;
        cvt_bf16_kernel<<<(unsigned)((n2 + 255) / 256), 256, 0, stream>>>(W_hh, Whh_bf, n2);
        long n3 = (long)MROWS * INF;
        cvt_feature_kernel<<<(unsigned)((n3 + 255) / 256), 256, 0, stream>>>(feature, Xbf);
        init_state_kernel<<<(BB * FEAT + 255) / 256, 256, 0, stream>>>(h2f, cbuf, h2bf);
    }

    // 2) hoisted input projection for all timesteps (one big WMMA GEMM)
    gemm_xw_kernel<<<dim3(MROWS / 32 / 8, G4 / 64), 256, 0, stream>>>(Xbf, Wih_bf, b_ih, Xg);

    // 3) sequential recurrence: 96 x (step GEMM + gating/heads)
    for (int t = 0; t < TT; ++t) {
        const float* xg_t = Xg + (size_t)t * BB * G4;
        step_gemm_kernel<<<FEAT / 16, 256, 0, stream>>>(h2bf, Whh_bf, b_hh, xg_t, cbuf, hnew);
        gate_heads_kernel<<<BB, 256, 0, stream>>>(h2f, hnew, W_use, b_use, W_pred, b_pred,
                                                  W_util, b_util, gumbel_u, t,
                                                  h2f, h2bf, pred_out, util_out);
    }
}
